// SelfAttentionResBlock_90134183674552
// MI455X (gfx1250) — matrix operand order repaired and checked by reading, not compiled, not run on hardware
//
#include <hip/hip_runtime.h>
#include <hip/hip_bf16.h>

// ---------------------------------------------------------------------------
// Self-attention residual block for MI455X (gfx1250), wave32 + WMMA f16.
//   x:(4,512,64,64) f32 -> GroupNorm -> QKV (1x1 conv GEMMs) -> softmax(QK^T/sqrt(C))V -> +x
// All matmuls use v_wmma_f32_16x16x32_f16 (f16 in, f32 accum).
// ---------------------------------------------------------------------------

typedef __attribute__((ext_vector_type(16))) _Float16 v16h;
typedef __attribute__((ext_vector_type(8)))  _Float16 v8h;
typedef __attribute__((ext_vector_type(8)))  float    v8f;

static constexpr int BB   = 4;      // batch
static constexpr int CC   = 512;    // channels
static constexpr int NN   = 4096;   // tokens = 64*64
static constexpr int GRP  = 32;     // groupnorm groups
static constexpr float EPSV = 1e-5f;

// ---------------- WMMA operand loaders (layouts per CDNA5 ISA) ---
// A 16x32 f16 (row-major source, row stride lda halfs):
//   lanes 0-15: M=lane, K in {0..7, 16..23}; lanes 16-31: M=lane-16, K in {8..15, 24..31}
__device__ __forceinline__ v16h load_a_16x32(const _Float16* base, int lda) {
  int lane = threadIdx.x & 31;
  const _Float16* p = base + (long)(lane & 15) * lda + (lane >> 4) * 8;
  union { v16h v; v8h h[2]; } u;
  u.h[0] = *(const v8h*)(p);        // K = half*8 .. +7
  u.h[1] = *(const v8h*)(p + 16);   // K = 16 + half*8 .. +7
  return u.v;
}

// B 32x16 f16 (column-contiguous source: element (k, col) at base[col*ldb + k]):
//   lanes 0-15: N=lane, K in {0..7, 16..23}; lanes 16-31: N=lane-16, K in {8..15, 24..31}
__device__ __forceinline__ v16h load_b_32x16(const _Float16* base, int ldb) {
  int lane = threadIdx.x & 31;
  const _Float16* p = base + (long)(lane & 15) * ldb + (lane >> 4) * 8;
  union { v16h v; v8h h[2]; } u;
  u.h[0] = *(const v8h*)(p);
  u.h[1] = *(const v8h*)(p + 16);
  return u.v;
}

__device__ __forceinline__ v8f wmma_f16(v16h a, v16h b, v8f c) {
  return __builtin_amdgcn_wmma_f32_16x16x32_f16(false, a, false, b, (short)0, c, false, false);
}

// ---------------- Kernel 1: GroupNorm -> f16 Xn, token-major (B,N,C) --------
__global__ void gn_kernel(const float* __restrict__ x,
                          const float* __restrict__ gamma,
                          const float* __restrict__ beta,
                          _Float16* __restrict__ Xn) {
  const int CG = CC / GRP;                       // 16 channels per group
  int b = blockIdx.x / GRP, g = blockIdx.x % GRP;
  const float* xg = x + (long)(b * CC + g * CG) * NN;
  int tid = threadIdx.x;

  float s = 0.f, s2 = 0.f;
  for (int i = tid; i < CG * NN; i += 256) { float v = xg[i]; s += v; s2 += v * v; }
  __shared__ float r1[256], r2[256];
  r1[tid] = s; r2[tid] = s2; __syncthreads();
  for (int ofs = 128; ofs > 0; ofs >>= 1) {
    if (tid < ofs) { r1[tid] += r1[tid + ofs]; r2[tid] += r2[tid + ofs]; }
    __syncthreads();
  }
  __shared__ float sMean, sRstd;
  if (tid == 0) {
    float inv = 1.0f / (float)(CG * NN);
    float mean = r1[0] * inv;
    float var  = r2[0] * inv - mean * mean;
    sMean = mean; sRstd = rsqrtf(var + EPSV);
  }
  __syncthreads();
  float mean = sMean, rstd = sRstd;

  int cg = tid & 15, c = g * CG + cg;
  float gm = gamma[c] * rstd;
  float bt = beta[c] - mean * gm;
  for (int n = (tid >> 4); n < NN; n += 16) {
    float v = xg[(long)cg * NN + n];
    Xn[(long)(b * NN + n) * CC + c] = (_Float16)(v * gm + bt);
  }
}

// ---------------- Kernel 2: f32 -> f16 weight conversion --------------------
__global__ void prep_w_kernel(const float* __restrict__ wq, const float* __restrict__ wk,
                              const float* __restrict__ wv, _Float16* __restrict__ Wh) {
  int i = blockIdx.x * 256 + threadIdx.x;
  if (i < CC * CC) {
    Wh[i]               = (_Float16)wq[i];
    Wh[CC * CC + i]     = (_Float16)wk[i];
    Wh[2 * CC * CC + i] = (_Float16)wv[i];
  }
}

// ---------------- Kernel 3: generic WMMA GEMM -------------------------------
// Out(M x Nn) = A(M x K) * B(K x Nn) + bias.  8 waves/block, wave = 16x64 tile.
__global__ void gemm_f16_kernel(const _Float16* __restrict__ A, int lda, long strideA,
                                const _Float16* __restrict__ Bc, int ldb, long strideB,
                                const float* __restrict__ bias, int biasPerN,
                                _Float16* __restrict__ Out, int ldo, long strideO,
                                int K) {
  int b    = blockIdx.z;
  int wid  = threadIdx.x >> 5;
  int lane = threadIdx.x & 31;
  int half = lane >> 4, l16 = lane & 15;
  int m0 = blockIdx.x * 16;
  int n0 = (blockIdx.y * 8 + wid) * 64;

  const _Float16* Ab = A  + b * strideA + (long)m0 * lda;
  const _Float16* Bb = Bc + b * strideB + (long)n0 * ldb;

  v8f acc[4] = {v8f{}, v8f{}, v8f{}, v8f{}};
#pragma unroll 4
  for (int k0 = 0; k0 < K; k0 += 32) {
    v16h a = load_a_16x32(Ab + k0, lda);
#pragma unroll
    for (int t = 0; t < 4; t++) {
      v16h bt = load_b_32x16(Bb + (long)(t * 16) * ldb + k0, ldb);
      acc[t] = wmma_f16(a, bt, acc[t]);
    }
  }
#pragma unroll
  for (int t = 0; t < 4; t++) {
    int n = n0 + t * 16 + l16;
#pragma unroll
    for (int i = 0; i < 8; i++) {
      int m = m0 + half * 8 + i;
      float bb = biasPerN ? bias[n] : bias[m];
      Out[b * strideO + (long)m * ldo + n] = (_Float16)(acc[t][i] + bb);
    }
  }
}

// ---------------- Kernel 4: flash attention + residual ----------------------
// Block (8 waves) owns 16 query rows. Per 64-column chunk:
//   - K-dim (512) split across the 8 waves for scores (2 k-steps each),
//     partial 16x64 tiles reduced in LDS via ds_add_f32 atomics
//   - online softmax on wave 0 (2 lanes per row, shfl_xor combine)
//   - each wave applies P x V for its private 64-channel output slice
__global__ void attn_kernel(const _Float16* __restrict__ Q,
                            const _Float16* __restrict__ Kt,
                            const _Float16* __restrict__ V,
                            const float* __restrict__ x,
                            float* __restrict__ out) {
  const float scale = 0.04419417382415922f;   // 1/sqrt(512)
  int b  = blockIdx.y;
  int m0 = blockIdx.x * 16;
  int tid = threadIdx.x, wid = tid >> 5, lane = tid & 31;
  int half = lane >> 4, l16 = lane & 15;

  __shared__ float  sS[16][64];                       // reduced score tile (4 KB)
  __shared__ __align__(16) _Float16 sP[16][64];       // probabilities (A operand)
  __shared__ float sScale[16], sM[16], sL[16];
  __shared__ float sT[8][16][17];                     // store-transpose staging

  int kbase = wid * 64;      // score K-slice per wave
  int cbase = wid * 64;      // PV channel slice per wave
  const _Float16* Qb = Q + (long)(b * NN + m0) * CC;

  // Q A-operands are chunk-invariant: load once, hold in VGPRs
  v16h aq0 = load_a_16x32(Qb + kbase, CC);
  v16h aq1 = load_a_16x32(Qb + kbase + 32, CC);

  v8f acc[4] = {v8f{}, v8f{}, v8f{}, v8f{}};
  if (tid < 16) { sM[tid] = -3.0e38f; sL[tid] = 0.f; }
  __syncthreads();

  for (int n0 = 0; n0 < NN; n0 += 64) {
    // prefetch next chunk's K rows / V lines (global_prefetch_b8)
    if (n0 + 64 < NN) {
      __builtin_prefetch(Kt + (long)(b * NN + n0 + 64 + lane) * CC + kbase, 0, 0);
      __builtin_prefetch(V + (long)(b * CC + cbase + l16) * NN + n0 + 64 + half * 32, 0, 0);
    }

    // zero score tile (1024 floats / 256 threads)
#pragma unroll
    for (int z = 0; z < 4; z++) ((float*)sS)[z * 256 + tid] = 0.f;
    __syncthreads();

    // partial scores over this wave's 64-wide K-slice (2 k-steps x 4 col-tiles)
    v8f sc[4] = {v8f{}, v8f{}, v8f{}, v8f{}};
#pragma unroll
    for (int nt = 0; nt < 4; nt++) {
      const _Float16* Kb = Kt + (long)(b * NN + n0 + nt * 16) * CC + kbase;
      v16h bk0 = load_b_32x16(Kb, CC);
      v16h bk1 = load_b_32x16(Kb + 32, CC);
      sc[nt] = wmma_f16(aq0, bk0, sc[nt]);
      sc[nt] = wmma_f16(aq1, bk1, sc[nt]);
    }
#pragma unroll
    for (int nt = 0; nt < 4; nt++)
#pragma unroll
      for (int i = 0; i < 8; i++)
        atomicAdd(&sS[half * 8 + i][nt * 16 + l16], sc[nt][i]);
    __syncthreads();

    // online softmax on wave 0: lanes r and r+16 split row r's 64 columns
    if (wid == 0) {
      int r = l16, hp = half;                 // hp selects columns hp*32..hp*32+31
      float mc = -3.0e38f;
      for (int j = 0; j < 32; j++) mc = fmaxf(mc, sS[r][hp * 32 + j] * scale);
      mc = fmaxf(mc, __shfl_xor(mc, 16, 32));
      float mNew = fmaxf(sM[r], mc);
      float corr = __expf(sM[r] - mNew);
      float sum = 0.f;
      for (int j = 0; j < 32; j++) {
        float p = __expf(sS[r][hp * 32 + j] * scale - mNew);
        sP[r][hp * 32 + j] = (_Float16)p;
        sum += p;
      }
      sum += __shfl_xor(sum, 16, 32);
      if (hp == 0) {
        sL[r] = sL[r] * corr + sum;
        sM[r] = mNew;
        sScale[r] = corr;
      }
    }
    __syncthreads();

    // rescale accumulators, then PV for this wave's 64 channels (K=64, 2 steps)
    float cr[8];
#pragma unroll
    for (int i = 0; i < 8; i++) cr[i] = sScale[half * 8 + i];
#pragma unroll
    for (int t = 0; t < 4; t++)
#pragma unroll
      for (int i = 0; i < 8; i++) acc[t][i] *= cr[i];

    v16h ap0 = load_a_16x32(&sP[0][0], 64);
    v16h ap1 = load_a_16x32(&sP[0][32], 64);
#pragma unroll
    for (int t = 0; t < 4; t++) {
      const _Float16* Vb = V + (long)(b * CC + cbase + t * 16) * NN + n0;
      v16h bv0 = load_b_32x16(Vb, NN);
      v16h bv1 = load_b_32x16(Vb + 32, NN);
      acc[t] = wmma_f16(ap0, bv0, acc[t]);
      acc[t] = wmma_f16(ap1, bv1, acc[t]);
    }
    __syncthreads();
  }

  // finalize: divide by denom, LDS-transpose, residual add, coalesced store
  float inv[8];
#pragma unroll
  for (int i = 0; i < 8; i++) inv[i] = 1.0f / sL[half * 8 + i];

  for (int t = 0; t < 4; t++) {
    __syncthreads();
#pragma unroll
    for (int i = 0; i < 8; i++) sT[wid][half * 8 + i][l16] = acc[t][i] * inv[i];
    __syncthreads();
    int r = l16;                       // token row after transpose
#pragma unroll
    for (int i = 0; i < 8; i++) {
      int cl = half * 8 + i;
      long idx = (long)(b * CC + cbase + t * 16 + cl) * NN + m0 + r;
      out[idx] = x[idx] + sT[wid][r][cl];
    }
  }
}

// ---------------- Host launcher ---------------------------------------------
extern "C" void kernel_launch(void* const* d_in, const int* in_sizes, int n_in,
                              void* d_out, int out_size, void* d_ws, size_t ws_size,
                              hipStream_t stream) {
  const float* x     = (const float*)d_in[0];
  const float* gamma = (const float*)d_in[1];
  const float* beta  = (const float*)d_in[2];
  const float* wq    = (const float*)d_in[3];
  const float* bq    = (const float*)d_in[4];
  const float* wk    = (const float*)d_in[5];
  const float* bk    = (const float*)d_in[6];
  const float* wv    = (const float*)d_in[7];
  const float* bv    = (const float*)d_in[8];
  float* out = (float*)d_out;

  char* ws = (char*)d_ws;
  const size_t SZ = (size_t)BB * NN * CC * sizeof(_Float16);   // 16 MB each
  _Float16* Xn = (_Float16*)(ws);
  _Float16* Qw = (_Float16*)(ws + SZ);
  _Float16* Kw = (_Float16*)(ws + 2 * SZ);
  _Float16* Vw = (_Float16*)(ws + 3 * SZ);
  _Float16* Wh = (_Float16*)(ws + 4 * SZ);                     // 3 * 512*512 f16

  // 1) GroupNorm -> Xn (token-major f16)
  gn_kernel<<<dim3(BB * GRP), 256, 0, stream>>>(x, gamma, beta, Xn);
  // 2) weights -> f16
  prep_w_kernel<<<dim3((CC * CC + 255) / 256), 256, 0, stream>>>(wq, wk, wv, Wh);

  const long sXn = (long)NN * CC;
  // 3) Q, K: (tokens x cout) = Xn (tokens x cin) * W^T, token-major output
  gemm_f16_kernel<<<dim3(NN / 16, 1, BB), 256, 0, stream>>>(
      Xn, CC, sXn, Wh, CC, 0, bq, 1, Qw, CC, sXn, CC);
  gemm_f16_kernel<<<dim3(NN / 16, 1, BB), 256, 0, stream>>>(
      Xn, CC, sXn, Wh + CC * CC, CC, 0, bk, 1, Kw, CC, sXn, CC);
  //    V: (cout x tokens) = W * Xn^T, channel-major output (B operand for PV)
  gemm_f16_kernel<<<dim3(CC / 16, NN / 512, BB), 256, 0, stream>>>(
      Wh + 2 * CC * CC, CC, 0, Xn, CC, sXn, bv, 0, Vw, NN, (long)CC * NN, CC);

  // 4) flash attention + residual
  attn_kernel<<<dim3(NN / 16, BB), 256, 0, stream>>>(Qw, Kw, Vw, x, out);
}
